// TrackHead_25907242729817
// MI455X (gfx1250) — matrix operand order repaired and checked by reading, not compiled, not run on hardware
//
#include <hip/hip_runtime.h>
#include <hip/hip_bf16.h>
#include <math.h>

typedef __attribute__((ext_vector_type(16))) __bf16 v16bf;
typedef __attribute__((ext_vector_type(8)))  float  v8f;

#define B_   32
#define C_   512
#define P_   576     // 24*24
#define N_   4096    // 64*64
#define WIMG 64
#define NTILES_I 36  // 576/16
#define NT   128     // columns per j-chunk (fallback path: 8 waves * 16)
#define NT2  256     // columns per j-chunk (fast path: 8 waves * 2 tiles * 16)
#define KC   128     // K chunk for fallback path
#define A_STRIDE 520 // fallback LDS pad
#define B_STRIDE 136 // fallback LDS pad

__device__ __forceinline__ unsigned short f2bf(float f) {
    unsigned int u = __float_as_uint(f);
    u += 0x7FFFu + ((u >> 16) & 1u);   // round-to-nearest-even
    return (unsigned short)(u >> 16);
}

// A fragment (16-bit A 16x32): per lane 16 consecutive K elems (two contiguous quads)
__device__ __forceinline__ v16bf load_frag_a(const unsigned short* base) {
    union { v16bf v; uint4 q[2]; } u;
    u.q[0] = *(const uint4*)(base);
    u.q[1] = *(const uint4*)(base + 8);
    return u.v;
}
// B fragment (16-bit B 32x16 layout): per lane 16 consecutive K elems (two contiguous quads)
__device__ __forceinline__ v16bf load_frag_b(const unsigned short* base) {
    union { v16bf v; uint4 q[2]; } u;
    u.q[0] = *(const uint4*)(base);
    u.q[1] = *(const uint4*)(base + 8);
    return u.v;
}

// -------- Kernel 1: inverse L2 norms over the channel dim --------
__global__ void norms_kernel(const float* __restrict__ x, float* __restrict__ inv,
                             int spatial, int total) {
    int idx = blockIdx.x * blockDim.x + threadIdx.x;
    if (idx >= total) return;
    int b = idx / spatial;
    int n = idx - b * spatial;
    const float* p = x + (size_t)b * C_ * spatial + n;
    float s = 0.f;
#pragma unroll 8
    for (int c = 0; c < C_; ++c) {
        float v = p[(size_t)c * spatial];
        s += v * v;
    }
    float nm = sqrtf(s);
    inv[idx] = 1.0f / fmaxf(nm, 1e-12f);
}

// -------- Kernel 2: normalize + convert + transpose: [B][C][S] f32 -> [B][S][C] bf16 --------
__global__ __launch_bounds__(256)
void cvt_transpose_kernel(const float* __restrict__ in, const float* __restrict__ inv,
                          unsigned short* __restrict__ out, int spatial) {
    const int b  = blockIdx.z;
    const int n0 = blockIdx.x * 32;
    const int c0 = blockIdx.y * 32;
    __shared__ unsigned short tile[32][33];
    const int tx = threadIdx.x;   // 0..31
    const int ty = threadIdx.y;   // 0..7
    const float s = inv[b * spatial + n0 + tx];
    const float* ip = in + (size_t)b * C_ * spatial + (size_t)c0 * spatial + n0 + tx;
#pragma unroll
    for (int i = 0; i < 32; i += 8)
        tile[i + ty][tx] = f2bf(ip[(size_t)(i + ty) * spatial] * s);
    __syncthreads();
    unsigned short* op = out + ((size_t)b * spatial + n0) * C_ + c0 + tx;
#pragma unroll
    for (int i = 0; i < 32; i += 8)
        op[(size_t)(i + ty) * C_] = tile[tx][i + ty];
}

// -------- Kernel 3 (fast path): WMMA GEMM, resident A, dual column tiles per wave --------
// srcB: [B][P][C] bf16 normalized, dstB: [B][N][C] bf16 normalized
// Scores are cosine similarities in [-1,1], so softmax needs no max-subtraction:
// Z = sum exp(s), Sx = sum exp(s)*cx, Sy = sum exp(s)*cy — all plain sums.
__global__ __launch_bounds__(256)
void affinity_bf16_kernel(const unsigned short* __restrict__ srcB,
                          const unsigned short* __restrict__ dstB,
                          float* __restrict__ partial) {
    const int b    = blockIdx.y;
    const int it   = blockIdx.x;          // i-tile 0..35
    const int i0   = it * 16;
    const int tid  = threadIdx.x;
    const int lane = tid & 31;
    const int w    = tid >> 5;            // wave 0..7
    const int n16  = lane & 15;
    const int half = lane >> 4;

    __shared__ float4 red[8 * 16];

    // Hoist the entire A tile (16 rows x K=512) into registers: 16 fragments = 128 VGPRs.
    const unsigned short* aBase =
        srcB + ((size_t)b * P_ + i0 + n16) * C_ + half * 16;
    v16bf aFrag[16];
#pragma unroll
    for (int s = 0; s < 16; ++s)
        aFrag[s] = load_frag_a(aBase + 32 * s);

    const int col0 = w * 32 + n16;        // tile0 column within j-chunk; tile1 = col0+16

    float Z0[8], Sx0[8], Sy0[8], Z1[8], Sx1[8], Sy1[8];
#pragma unroll
    for (int r = 0; r < 8; ++r) {
        Z0[r] = 0.f; Sx0[r] = 0.f; Sy0[r] = 0.f;
        Z1[r] = 0.f; Sx1[r] = 0.f; Sy1[r] = 0.f;
    }

    for (int j0 = 0; j0 < N_; j0 += NT2) {
        const unsigned short* bBase0 =
            dstB + ((size_t)b * N_ + j0 + col0) * C_ + half * 16;
        const unsigned short* bBase1 = bBase0 + (size_t)16 * C_;
        __builtin_prefetch(bBase0 + (size_t)NT2 * C_, 0, 1);   // next chunk's column
        v8f acc0 = {0.f, 0.f, 0.f, 0.f, 0.f, 0.f, 0.f, 0.f};
        v8f acc1 = {0.f, 0.f, 0.f, 0.f, 0.f, 0.f, 0.f, 0.f};
#pragma unroll
        for (int s = 0; s < 16; ++s) {
            v16bf bf0 = load_frag_b(bBase0 + 32 * s);
            v16bf bf1 = load_frag_b(bBase1 + 32 * s);
            acc0 = __builtin_amdgcn_wmma_f32_16x16x32_bf16(
                    false, aFrag[s], false, bf0, (short)0, acc0, false, false);
            acc1 = __builtin_amdgcn_wmma_f32_16x16x32_bf16(
                    false, aFrag[s], false, bf1, (short)0, acc1, false, false);
        }
        const int jc0 = j0 + col0;                    // low 6 bits <= 47, +16 never wraps
        const float cx0 = (float)(jc0 & (WIMG - 1)) + 0.5f;
        const float cx1 = cx0 + 16.f;
        const float cy  = (float)(jc0 >> 6) + 0.5f;   // same row for both tiles
#pragma unroll
        for (int r = 0; r < 8; ++r) {
            float e0 = __expf(acc0[r]);
            float e1 = __expf(acc1[r]);
            Z0[r]  += e0;        Z1[r]  += e1;
            Sx0[r] += e0 * cx0;  Sx1[r] += e1 * cx1;
            Sy0[r] += e0 * cy;   Sy1[r] += e1 * cy;
        }
    }

    // combine the two tiles (same rows, disjoint columns) — plain sums
#pragma unroll
    for (int r = 0; r < 8; ++r) {
        Z0[r] += Z1[r]; Sx0[r] += Sx1[r]; Sy0[r] += Sy1[r];
    }
    // reduce across the 16 lanes of each half-wave
#pragma unroll
    for (int mask = 1; mask <= 8; mask <<= 1) {
#pragma unroll
        for (int r = 0; r < 8; ++r) {
            Z0[r]  += __shfl_xor(Z0[r],  mask, 32);
            Sx0[r] += __shfl_xor(Sx0[r], mask, 32);
            Sy0[r] += __shfl_xor(Sy0[r], mask, 32);
        }
    }
    if ((lane & 15) == 0) {
#pragma unroll
        for (int r = 0; r < 8; ++r)
            red[w * 16 + half * 8 + r] = make_float4(Z0[r], Sx0[r], Sy0[r], 0.f);
    }
    __syncthreads();

    if (tid < 16) {
        const int row = tid;
        float Z = 0.f, Sx = 0.f, Sy = 0.f;
#pragma unroll
        for (int ww = 0; ww < 8; ++ww) {
            float4 u = red[ww * 16 + row];
            Z += u.x; Sx += u.y; Sy += u.z;
        }
        float px = Sx / Z;
        float py = Sy / Z;
#pragma unroll
        for (int mask = 1; mask <= 8; mask <<= 1) {
            px += __shfl_xor(px, mask, 32);
            py += __shfl_xor(py, mask, 32);
        }
        if (tid == 0) {
            partial[(b * NTILES_I + it) * 2 + 0] = px;
            partial[(b * NTILES_I + it) * 2 + 1] = py;
        }
    }
}

// -------- Fallback: fused on-the-fly conversion kernel (small ws) --------
__global__ __launch_bounds__(256)
void affinity_fused_kernel(const float* __restrict__ patch_x, const float* __restrict__ x,
                           const float* __restrict__ inv_src, const float* __restrict__ inv_dst,
                           float* __restrict__ partial) {
    const int b    = blockIdx.y;
    const int it   = blockIdx.x;
    const int i0   = it * 16;
    const int tid  = threadIdx.x;
    const int lane = tid & 31;
    const int w    = tid >> 5;
    const int n16  = lane & 15;
    const int half = lane >> 4;

    __shared__ __align__(16) unsigned short aT[16 * A_STRIDE];
    __shared__ __align__(16) unsigned short bT[NT * B_STRIDE];
    __shared__ float4 red[8 * 16];

    {
        const int m = tid & 15;
        const float is = inv_src[b * P_ + i0 + m];
        const float* src = patch_x + (size_t)b * C_ * P_ + (i0 + m);
        for (int c = tid >> 4; c < C_; c += 16)
            aT[m * A_STRIDE + c] = f2bf(src[(size_t)c * P_] * is);
    }

    float Zs[8], Sx[8], Sy[8];
#pragma unroll
    for (int r = 0; r < 8; ++r) { Zs[r] = 0.f; Sx[r] = 0.f; Sy[r] = 0.f; }

    const int jj = tid & (NT - 1);
    const int kg = tid >> 7;
    const int myCol = w * 16 + n16;

    for (int j0 = 0; j0 < N_; j0 += NT) {
        v8f acc = {0.f, 0.f, 0.f, 0.f, 0.f, 0.f, 0.f, 0.f};
        const float idn = inv_dst[b * N_ + j0 + jj];
        const float* xcol = x + (size_t)b * C_ * N_ + (j0 + jj);
        for (int kc = 0; kc < C_; kc += KC) {
            __syncthreads();
            for (int k = kg * 2; k < KC; k += 4) {
                float v0 = xcol[(size_t)(kc + k)     * N_] * idn;
                float v1 = xcol[(size_t)(kc + k + 1) * N_] * idn;
                unsigned int pk = (unsigned int)f2bf(v0) | ((unsigned int)f2bf(v1) << 16);
                *(unsigned int*)&bT[jj * B_STRIDE + k] = pk;
            }
            __syncthreads();
#pragma unroll
            for (int s = 0; s < 4; ++s) {
                const int kb = s * 32;
                v16bf af = load_frag_a(&aT[n16 * A_STRIDE + kc + kb + half * 16]);
                v16bf bf = load_frag_b(&bT[myCol * B_STRIDE + kb + half * 16]);
                acc = __builtin_amdgcn_wmma_f32_16x16x32_bf16(
                        false, af, false, bf, (short)0, acc, false, false);
            }
        }
        const int jc = j0 + myCol;
        const float cx = (float)(jc & (WIMG - 1)) + 0.5f;
        const float cy = (float)(jc >> 6) + 0.5f;
#pragma unroll
        for (int r = 0; r < 8; ++r) {
            float e = __expf(acc[r]);
            Zs[r] += e;
            Sx[r] += e * cx;
            Sy[r] += e * cy;
        }
    }
#pragma unroll
    for (int mask = 1; mask <= 8; mask <<= 1) {
#pragma unroll
        for (int r = 0; r < 8; ++r) {
            Zs[r] += __shfl_xor(Zs[r], mask, 32);
            Sx[r] += __shfl_xor(Sx[r], mask, 32);
            Sy[r] += __shfl_xor(Sy[r], mask, 32);
        }
    }
    if ((lane & 15) == 0) {
#pragma unroll
        for (int r = 0; r < 8; ++r)
            red[w * 16 + half * 8 + r] = make_float4(Zs[r], Sx[r], Sy[r], 0.f);
    }
    __syncthreads();
    if (tid < 16) {
        const int row = tid;
        float Z = 0.f, SxT = 0.f, SyT = 0.f;
#pragma unroll
        for (int ww = 0; ww < 8; ++ww) {
            float4 u = red[ww * 16 + row];
            Z += u.x; SxT += u.y; SyT += u.z;
        }
        float px = SxT / Z;
        float py = SyT / Z;
#pragma unroll
        for (int mask = 1; mask <= 8; mask <<= 1) {
            px += __shfl_xor(px, mask, 32);
            py += __shfl_xor(py, mask, 32);
        }
        if (tid == 0) {
            partial[(b * NTILES_I + it) * 2 + 0] = px;
            partial[(b * NTILES_I + it) * 2 + 1] = py;
        }
    }
}

// -------- Kernel 4: reduce tile partials -> bbox --------
__global__ void bbox_kernel(const float* __restrict__ partial, float* __restrict__ out) {
    int b = threadIdx.x;
    if (b < B_) {
        float sx = 0.f, sy = 0.f;
        for (int i = 0; i < NTILES_I; ++i) {
            sx += partial[(b * NTILES_I + i) * 2 + 0];
            sy += partial[(b * NTILES_I + i) * 2 + 1];
        }
        float cx = sx / (float)P_;
        float cy = sy / (float)P_;
        float l  = fmaxf(cx - 12.f, 0.f);
        float t  = fmaxf(cy - 12.f, 0.f);
        float r  = fminf(l + 24.f, 64.f);
        float bo = fminf(t + 24.f, 64.f);
        out[b * 4 + 0] = l;
        out[b * 4 + 1] = t;
        out[b * 4 + 2] = r;
        out[b * 4 + 3] = bo;
    }
}

extern "C" void kernel_launch(void* const* d_in, const int* in_sizes, int n_in,
                              void* d_out, int out_size, void* d_ws, size_t ws_size,
                              hipStream_t stream) {
    const float* patch_x = (const float*)d_in[0];   // [32,512,24,24]
    const float* x       = (const float*)d_in[1];   // [32,512,64,64]

    char* ws = (char*)d_ws;
    float* inv_dst = (float*)ws;                                   // B*N f32
    float* inv_src = inv_dst + (size_t)B_ * N_;                    // B*P f32
    float* partial = inv_src + (size_t)B_ * P_;                    // B*36*2 f32
    size_t small_bytes = ((size_t)B_ * N_ + (size_t)B_ * P_ + (size_t)B_ * NTILES_I * 2) * 4;
    size_t off = (small_bytes + 255) & ~(size_t)255;
    unsigned short* srcB = (unsigned short*)(ws + off);            // B*P*C bf16
    size_t src_bytes = (size_t)B_ * P_ * C_ * 2;
    unsigned short* dstB = (unsigned short*)(ws + off + ((src_bytes + 255) & ~(size_t)255));
    size_t dst_bytes = (size_t)B_ * N_ * C_ * 2;
    size_t need = off + ((src_bytes + 255) & ~(size_t)255) + dst_bytes;

    norms_kernel<<<(B_ * N_ + 255) / 256, 256, 0, stream>>>(x, inv_dst, N_, B_ * N_);
    norms_kernel<<<(B_ * P_ + 255) / 256, 256, 0, stream>>>(patch_x, inv_src, P_, B_ * P_);

    dim3 grid(NTILES_I, B_);
    if (ws_size >= need) {
        dim3 tb(32, 8);
        dim3 gd(N_ / 32, C_ / 32, B_);
        cvt_transpose_kernel<<<gd, tb, 0, stream>>>(x, inv_dst, dstB, N_);
        dim3 gs(P_ / 32, C_ / 32, B_);
        cvt_transpose_kernel<<<gs, tb, 0, stream>>>(patch_x, inv_src, srcB, P_);
        affinity_bf16_kernel<<<grid, 256, 0, stream>>>(srcB, dstB, partial);
    } else {
        affinity_fused_kernel<<<grid, 256, 0, stream>>>(patch_x, x, inv_src, inv_dst, partial);
    }

    bbox_kernel<<<1, 64, 0, stream>>>(partial, (float*)d_out);
}
